// SpeMambaBlock_50173807952195
// MI455X (gfx1250) — compile-verified, matrix operand order checked by reading, not run on hardware
//
#include <hip/hip_runtime.h>
#include <hip/hip_bf16.h>
#include <stdint.h>

typedef __attribute__((ext_vector_type(16))) _Float16 v16h;
typedef __attribute__((ext_vector_type(8)))  float    v8f;

#define HD __device__ __forceinline__

// ---------------- problem constants ----------------
#define BB   2
#define CC   1024
#define HH   64
#define WWd  64
#define TT   8
#define GG   128     // d_model
#define DIN  256     // d_inner
#define DS   16
#define DCV  4
#define DTR  8
#define NPIX (BB*HH*WWd)     // 8192
#define MROWS (NPIX*TT)      // 65536
#define NXP  48              // x_proj out padded 40 -> 48
#define HWSZ (HH*WWd)        // 4096

// ---------------- workspace layout (bytes, 256B aligned) ----------------
static constexpr size_t OFF_A0   = 0;                       // 65536*128 f16  = 16 MB
static constexpr size_t OFF_W1   = OFF_A0   + 16777216;     // 512*128  f16
static constexpr size_t OFF_WXP  = OFF_W1   + 131072;       // 48*256   f16
static constexpr size_t OFF_WOUT = OFF_WXP  + 24576;        // 128*256  f16
static constexpr size_t OFF_UCF  = OFF_WOUT + 65536;        // 65536*256 f32 (conv+silu u)
static constexpr size_t OFF_UCH  = OFF_UCF  + 67108864;     // 65536*256 f16
static constexpr size_t OFF_ZS   = OFF_UCH  + 33554432;     // 65536*256 f32 (silu(z))
static constexpr size_t OFF_XDBL = OFF_ZS   + 67108864;     // 65536*48  f32
static constexpr size_t OFF_DT   = OFF_XDBL + 12582912;     // 65536*256 f32 (softplus dt)
static constexpr size_t OFF_YH   = OFF_DT   + 67108864;     // 65536*256 f16 (gated y)
static constexpr size_t OFF_YM   = OFF_YH   + 33554432;     // 65536*128 f32 (out_proj)
static constexpr size_t OFF_GN   = OFF_YM   + 33554432;     // 16 f32 stats

// ---------------- helpers ----------------
HD float sigm(float x) { return 1.0f / (1.0f + __expf(-x)); }

// A/B fragment load from an LDS tile stored [row-or-col][k], k contiguous,
// row stride = 40 halves (80B, 16B aligned -> two ds_load_b128 per frag).
// Lane l: rc = l&15, half = l>>4; element e maps to k = (e>>3)*16 + half*8 + (e&7).
HD v16h ldfrag(const _Float16* base, int rc, int half, int stride) {
  const _Float16* p = base + rc * stride + half * 8;
  v16h f;
#pragma unroll
  for (int i = 0; i < 8; ++i) f[i] = p[i];
#pragma unroll
  for (int i = 0; i < 8; ++i) f[8 + i] = p[16 + i];
  return f;
}

HD v8f wmma16(v16h a, v16h b, v8f c) {
  return __builtin_amdgcn_wmma_f32_16x16x32_f16(false, a, false, b, (short)0, c, false, false);
}

// ---------------- kernel 1: cast weights to f16 (x_proj padded to 48 rows) ----------------
__global__ void k_wconv(const float* w1, const float* wxp, const float* wout,
                        _Float16* w1h, _Float16* wxph, _Float16* wouth) {
  int id = blockIdx.x * 256 + threadIdx.x;
  if (id < 512 * 128) {
    w1h[id] = (_Float16)w1[id];
  } else if (id < 512 * 128 + NXP * 256) {
    int j = id - 512 * 128;
    int n = j >> 8, k = j & 255;
    float v = (n < 40) ? wxp[n * 256 + k] : 0.0f;
    wxph[j] = (_Float16)v;
  } else if (id < 512 * 128 + NXP * 256 + 128 * 256) {
    int j = id - (512 * 128 + NXP * 256);
    wouth[j] = (_Float16)wout[j];
  }
}

// ---------------- kernel 2: x (B,C,H,W) -> A0 f16 (M=65536, K=128) ----------------
// row r = p*8 + t with p = ((b*H+h)*W+w); col g;  source c = t*128+g
__global__ void k_prep(const float* x, _Float16* A0) {
  int id = blockIdx.x * 256 + threadIdx.x;     // 8388608 total
  int b = id >> 22;
  int c = (id >> 12) & 1023;
  int hw = id & 4095;
  int t = c >> 7, g = c & 127;
  int p = b * HWSZ + hw;
  A0[(size_t)(p * TT + t) * GG + g] = (_Float16)x[id];
}

// ---------------- kernel 3: in_proj GEMM + fused conv/SiLU (u) and SiLU (z) ----------------
// C tile 64(M) x 128(N); 8 waves, each 32x32 (2x2 WMMA). grid = (1024, 4)
__global__ __launch_bounds__(256) void k_gemm_inproj(
    const _Float16* __restrict__ A0, const _Float16* __restrict__ W1h,
    const float* __restrict__ conv_w, const float* __restrict__ conv_b,
    float* __restrict__ ucF, _Float16* __restrict__ ucH, float* __restrict__ zS) {
  __shared__ __align__(16) _Float16 As[64][40];
  __shared__ __align__(16) _Float16 Bs[128][40];
  __shared__ float Ot[64][128];

  const int tid = threadIdx.x;
  const int lane = tid & 31, wid = tid >> 5;
  const int half = lane >> 4, rc = lane & 15;
  const int mBase = blockIdx.x * 64;
  const int nBase = blockIdx.y * 128;
  const int m0 = (wid & 1) * 32, n0 = (wid >> 1) * 32;

  v8f acc[2][2];
#pragma unroll
  for (int i = 0; i < 2; ++i)
#pragma unroll
    for (int j = 0; j < 2; ++j) acc[i][j] = (v8f)(0.0f);

  for (int kt = 0; kt < 4; ++kt) {
    // A tile 64x32: one float4 (8 halves) per thread
    {
      int row = tid >> 2, kk = (tid & 3) * 8;
      *(float4*)&As[row][kk] =
          *(const float4*)&A0[(size_t)(mBase + row) * GG + kt * 32 + kk];
      if (kt + 1 < 4)
        __builtin_prefetch(&A0[(size_t)(mBase + row) * GG + (kt + 1) * 32 + kk], 0, 1);
    }
    // B tile 128x32: two float4 per thread
    {
      int n = tid >> 1, kk = (tid & 1) * 16;
      const _Float16* src = &W1h[(size_t)(nBase + n) * GG + kt * 32 + kk];
      *(float4*)&Bs[n][kk] = *(const float4*)src;
      *(float4*)&Bs[n][kk + 8] = *(const float4*)(src + 8);
    }
    __syncthreads();
    v16h af[2], bf[2];
#pragma unroll
    for (int i = 0; i < 2; ++i) af[i] = ldfrag(&As[m0 + i * 16][0], rc, half, 40);
#pragma unroll
    for (int j = 0; j < 2; ++j) bf[j] = ldfrag(&Bs[n0 + j * 16][0], rc, half, 40);
#pragma unroll
    for (int i = 0; i < 2; ++i)
#pragma unroll
      for (int j = 0; j < 2; ++j) acc[i][j] = wmma16(af[i], bf[j], acc[i][j]);
    __syncthreads();
  }

  // spill accumulators to LDS: lane l -> row v + 8*half, col l&15
#pragma unroll
  for (int i = 0; i < 2; ++i)
#pragma unroll
    for (int j = 0; j < 2; ++j)
#pragma unroll
      for (int v = 0; v < 8; ++v)
        Ot[m0 + i * 16 + v + 8 * half][n0 + j * 16 + rc] = acc[i][j][v];
  __syncthreads();

  if (nBase < 256) {
    // u region: fused depthwise causal conv (DC=4, left pad 3) + bias + SiLU.
    // 8 sequences x 128 channels = 1024 tasks, 4 per thread.
#pragma unroll
    for (int q = 0; q < 4; ++q) {
      int idx = tid + q * 256;
      int sl = idx >> 7;           // local sequence 0..7
      int cl = idx & 127;          // local channel
      int cg = nBase + cl;         // global u channel 0..255
      float w0 = conv_w[cg * DCV + 0], w1 = conv_w[cg * DCV + 1];
      float w2 = conv_w[cg * DCV + 2], w3 = conv_w[cg * DCV + 3];
      float bias = conv_b[cg];
      float u[TT];
#pragma unroll
      for (int t = 0; t < TT; ++t) u[t] = Ot[sl * TT + t][cl];
#pragma unroll
      for (int t = 0; t < TT; ++t) {
        float s = w3 * u[t];
        if (t >= 1) s += w2 * u[t - 1];
        if (t >= 2) s += w1 * u[t - 2];
        if (t >= 3) s += w0 * u[t - 3];
        s += bias;
        float act = s * sigm(s);
        size_t r = (size_t)(mBase + sl * TT + t);
        ucF[r * DIN + cg] = act;
        ucH[r * DIN + cg] = (_Float16)act;
      }
    }
  } else {
    // z region: store silu(z)
#pragma unroll
    for (int q = 0; q < 32; ++q) {
      int e = tid + q * 256;
      int row = e >> 7, cl = e & 127;
      float z = Ot[row][cl];
      zS[(size_t)(mBase + row) * DIN + (nBase - 256 + cl)] = z * sigm(z);
    }
  }
}

// ---------------- kernel 4: x_proj GEMM (K=256, N=48) ----------------
// block tile 128(M) x 48(N); 8 waves each one 16-row strip x 3 tiles. grid = 512
__global__ __launch_bounds__(256) void k_gemm_xproj(
    const _Float16* __restrict__ ucH, const _Float16* __restrict__ Wxph,
    float* __restrict__ xdbl) {
  __shared__ __align__(16) _Float16 As[128][40];
  __shared__ __align__(16) _Float16 Bs[48][40];
  const int tid = threadIdx.x;
  const int lane = tid & 31, wid = tid >> 5;
  const int half = lane >> 4, rc = lane & 15;
  const int mBase = blockIdx.x * 128;
  const int m0 = wid * 16;

  v8f acc[3];
#pragma unroll
  for (int j = 0; j < 3; ++j) acc[j] = (v8f)(0.0f);

  for (int kt = 0; kt < 8; ++kt) {
    {
      int row = tid >> 1, kk = (tid & 1) * 16;
      const _Float16* src = &ucH[(size_t)(mBase + row) * DIN + kt * 32 + kk];
      *(float4*)&As[row][kk] = *(const float4*)src;
      *(float4*)&As[row][kk + 8] = *(const float4*)(src + 8);
    }
    if (tid < 192) {
      int n = tid >> 2, kk = (tid & 3) * 8;
      *(float4*)&Bs[n][kk] =
          *(const float4*)&Wxph[(size_t)n * DIN + kt * 32 + kk];
    }
    __syncthreads();
    v16h a = ldfrag(&As[m0][0], rc, half, 40);
#pragma unroll
    for (int j = 0; j < 3; ++j) {
      v16h b = ldfrag(&Bs[j * 16][0], rc, half, 40);
      acc[j] = wmma16(a, b, acc[j]);
    }
    __syncthreads();
  }
#pragma unroll
  for (int j = 0; j < 3; ++j)
#pragma unroll
    for (int v = 0; v < 8; ++v) {
      int row = mBase + m0 + v + 8 * half;
      xdbl[(size_t)row * NXP + j * 16 + rc] = acc[j][v];
    }
}

// ---------------- kernel 5: dt_proj (K=8) + softplus ----------------
__global__ void k_dtproj(const float* __restrict__ xdbl,
                         const float* __restrict__ dtw, const float* __restrict__ dtb,
                         float* __restrict__ dtbuf) {
  __shared__ float dtr[DTR];
  int r = blockIdx.x, d = threadIdx.x;
  if (d < DTR) dtr[d] = xdbl[(size_t)r * NXP + d];
  __syncthreads();
  float s = dtb[d];
#pragma unroll
  for (int j = 0; j < DTR; ++j) s += dtr[j] * dtw[d * DTR + j];
  float sp = (s > 20.0f) ? s : __logf(1.0f + __expf(s));
  dtbuf[(size_t)r * DIN + d] = sp;
}

// ---------------- kernel 6: selective scan + D skip + gating ----------------
// one block per pixel; thread d owns h[16] in registers.
__global__ __launch_bounds__(256) void k_scan(
    const float* __restrict__ xdbl, const float* __restrict__ dtbuf,
    const float* __restrict__ ucF, const float* __restrict__ zS,
    const float* __restrict__ A_log, const float* __restrict__ Dv,
    _Float16* __restrict__ yH) {
  __shared__ float Bsh[TT][DS], Csh[TT][DS];
  const int p = blockIdx.x, d = threadIdx.x;
  if (d < 128) {
    int t = d >> 4, s = d & 15;
    Bsh[t][s] = xdbl[(size_t)(p * TT + t) * NXP + DTR + s];
  } else {
    int e = d - 128;
    int t = e >> 4, s = e & 15;
    Csh[t][s] = xdbl[(size_t)(p * TT + t) * NXP + DTR + DS + s];
  }
  __syncthreads();

  float An[DS];
#pragma unroll
  for (int s = 0; s < DS; ++s) An[s] = -__expf(A_log[d * DS + s]);
  float h[DS];
#pragma unroll
  for (int s = 0; s < DS; ++s) h[s] = 0.0f;
  const float Dd = Dv[d];

#pragma unroll
  for (int t = 0; t < TT; ++t) {
    size_t r = (size_t)(p * TT + t);
    float dt = dtbuf[r * DIN + d];
    float u = ucF[r * DIN + d];
    float du = dt * u;
    float y = 0.0f;
#pragma unroll
    for (int s = 0; s < DS; ++s) {
      float dA = __expf(dt * An[s]);
      h[s] = dA * h[s] + du * Bsh[t][s];
      y += h[s] * Csh[t][s];
    }
    float yo = (y + u * Dd) * zS[r * DIN + d];
    yH[r * DIN + d] = (_Float16)yo;
  }
}

// ---------------- kernel 7: out_proj GEMM (K=256, N=128) ----------------
// block tile 64x128, 8 waves of 32x32. grid = 1024
__global__ __launch_bounds__(256) void k_gemm_outproj(
    const _Float16* __restrict__ yH, const _Float16* __restrict__ Wouth,
    float* __restrict__ ym) {
  __shared__ __align__(16) _Float16 As[64][40];
  __shared__ __align__(16) _Float16 Bs[128][40];
  const int tid = threadIdx.x;
  const int lane = tid & 31, wid = tid >> 5;
  const int half = lane >> 4, rc = lane & 15;
  const int mBase = blockIdx.x * 64;
  const int m0 = (wid & 1) * 32, n0 = (wid >> 1) * 32;

  v8f acc[2][2];
#pragma unroll
  for (int i = 0; i < 2; ++i)
#pragma unroll
    for (int j = 0; j < 2; ++j) acc[i][j] = (v8f)(0.0f);

  for (int kt = 0; kt < 8; ++kt) {
    {
      int row = tid >> 2, kk = (tid & 3) * 8;
      *(float4*)&As[row][kk] =
          *(const float4*)&yH[(size_t)(mBase + row) * DIN + kt * 32 + kk];
      if (kt + 1 < 8)
        __builtin_prefetch(&yH[(size_t)(mBase + row) * DIN + (kt + 1) * 32 + kk], 0, 1);
    }
    {
      int n = tid >> 1, kk = (tid & 1) * 16;
      const _Float16* src = &Wouth[(size_t)n * DIN + kt * 32 + kk];
      *(float4*)&Bs[n][kk] = *(const float4*)src;
      *(float4*)&Bs[n][kk + 8] = *(const float4*)(src + 8);
    }
    __syncthreads();
    v16h af[2], bf[2];
#pragma unroll
    for (int i = 0; i < 2; ++i) af[i] = ldfrag(&As[m0 + i * 16][0], rc, half, 40);
#pragma unroll
    for (int j = 0; j < 2; ++j) bf[j] = ldfrag(&Bs[n0 + j * 16][0], rc, half, 40);
#pragma unroll
    for (int i = 0; i < 2; ++i)
#pragma unroll
      for (int j = 0; j < 2; ++j) acc[i][j] = wmma16(af[i], bf[j], acc[i][j]);
    __syncthreads();
  }
#pragma unroll
  for (int i = 0; i < 2; ++i)
#pragma unroll
    for (int j = 0; j < 2; ++j)
#pragma unroll
      for (int v = 0; v < 8; ++v) {
        int row = mBase + m0 + i * 16 + v + 8 * half;
        ym[(size_t)row * GG + n0 + j * 16 + rc] = acc[i][j][v];
      }
}

// ---------------- kernels 8-10: GroupNorm stats + apply + SiLU + residual ----------------
__global__ void k_gn_init(float* gn) {
  if (threadIdx.x < 16) gn[threadIdx.x] = 0.0f;
}

__global__ __launch_bounds__(256) void k_gn_stats(const float* __restrict__ ym,
                                                  float* __restrict__ gn) {
  __shared__ float rs[256], rs2[256];
  int gidx = blockIdx.x >> 5;        // 0..7  (b*4 + group)
  int slice = blockIdx.x & 31;       // 32 pixel slices of 128
  int b = gidx >> 2, gi = gidx & 3;
  int tid = threadIdx.x;
  float s = 0.0f, s2 = 0.0f;
  for (int e = tid; e < 32768; e += 256) {
    int pl = e >> 8;
    int tl = (e >> 7) & 1;
    int g = e & 127;
    int p = b * HWSZ + slice * 128 + pl;
    int t = gi * 2 + tl;
    float v = ym[(size_t)(p * TT + t) * GG + g];
    s += v;
    s2 += v * v;
  }
  rs[tid] = s; rs2[tid] = s2;
  __syncthreads();
  for (int off = 128; off > 0; off >>= 1) {
    if (tid < off) { rs[tid] += rs[tid + off]; rs2[tid] += rs2[tid + off]; }
    __syncthreads();
  }
  if (tid == 0) {
    atomicAdd(&gn[gidx * 2], rs[0]);
    atomicAdd(&gn[gidx * 2 + 1], rs2[0]);
  }
}

__global__ void k_gn_apply(const float* __restrict__ x, const float* __restrict__ ym,
                           const float* __restrict__ gn,
                           const float* __restrict__ gnw, const float* __restrict__ gnb,
                           float* __restrict__ out) {
  int id = blockIdx.x * 256 + threadIdx.x;
  int b = id >> 22;
  int c = (id >> 12) & 1023;
  int hw = id & 4095;
  int gidx = b * 4 + (c >> 8);
  const float n = 1048576.0f;
  float mu = gn[gidx * 2] / n;
  float var = gn[gidx * 2 + 1] / n - mu * mu;
  float inv = rsqrtf(var + 1e-5f);
  int t = c >> 7, g = c & 127;
  int p = b * HWSZ + hw;
  float v = ym[(size_t)(p * TT + t) * GG + g];
  float xn = (v - mu) * inv * gnw[c] + gnb[c];
  out[id] = x[id] + xn * sigm(xn);
}

// ---------------- launcher ----------------
extern "C" void kernel_launch(void* const* d_in, const int* in_sizes, int n_in,
                              void* d_out, int out_size, void* d_ws, size_t ws_size,
                              hipStream_t stream) {
  const float* x        = (const float*)d_in[0];
  const float* in_projw = (const float*)d_in[1];
  const float* conv_w   = (const float*)d_in[2];
  const float* conv_b   = (const float*)d_in[3];
  const float* x_projw  = (const float*)d_in[4];
  const float* dt_projw = (const float*)d_in[5];
  const float* dt_projb = (const float*)d_in[6];
  const float* A_log    = (const float*)d_in[7];
  const float* Dv       = (const float*)d_in[8];
  const float* out_projw= (const float*)d_in[9];
  const float* gnw      = (const float*)d_in[10];
  const float* gnb      = (const float*)d_in[11];
  float* out = (float*)d_out;

  char* ws = (char*)d_ws;
  _Float16* A0    = (_Float16*)(ws + OFF_A0);
  _Float16* W1h   = (_Float16*)(ws + OFF_W1);
  _Float16* Wxph  = (_Float16*)(ws + OFF_WXP);
  _Float16* Wouth = (_Float16*)(ws + OFF_WOUT);
  float*    ucF   = (float*)   (ws + OFF_UCF);
  _Float16* ucH   = (_Float16*)(ws + OFF_UCH);
  float*    zS    = (float*)   (ws + OFF_ZS);
  float*    xdbl  = (float*)   (ws + OFF_XDBL);
  float*    dtbuf = (float*)   (ws + OFF_DT);
  _Float16* yH    = (_Float16*)(ws + OFF_YH);
  float*    ym    = (float*)   (ws + OFF_YM);
  float*    gn    = (float*)   (ws + OFF_GN);

  k_wconv<<<432, 256, 0, stream>>>(in_projw, x_projw, out_projw, W1h, Wxph, Wouth);
  k_prep<<<32768, 256, 0, stream>>>(x, A0);
  k_gemm_inproj<<<dim3(1024, 4), 256, 0, stream>>>(A0, W1h, conv_w, conv_b, ucF, ucH, zS);
  k_gemm_xproj<<<512, 256, 0, stream>>>(ucH, Wxph, xdbl);
  k_dtproj<<<65536, 256, 0, stream>>>(xdbl, dt_projw, dt_projb, dtbuf);
  k_scan<<<NPIX, 256, 0, stream>>>(xdbl, dtbuf, ucF, zS, A_log, Dv, yH);
  k_gemm_outproj<<<1024, 256, 0, stream>>>(yH, Wouth, ym);
  k_gn_init<<<1, 64, 0, stream>>>(gn);
  k_gn_stats<<<256, 256, 0, stream>>>(ym, gn);
  k_gn_apply<<<32768, 256, 0, stream>>>(x, ym, gn, gnw, gnb, out);
}
